// QuantizedSSM_74938589381381
// MI455X (gfx1250) — matrix operand order repaired and checked by reading, not compiled, not run on hardware
//
#include <hip/hip_runtime.h>

// ---------------------------------------------------------------------------
// Types
// ---------------------------------------------------------------------------
typedef __attribute__((ext_vector_type(16))) __bf16         v16bf;
typedef __attribute__((ext_vector_type(8)))  float          v8f;
typedef __attribute__((ext_vector_type(4)))  unsigned int   v4u;
typedef __attribute__((ext_vector_type(4)))  unsigned short v4us;
typedef __attribute__((ext_vector_type(4)))  float          v4f;
typedef int v4i_async __attribute__((vector_size(4 * sizeof(int))));  // builtin's pointee type

struct Frag {          // 32 bytes == one v16bf WMMA operand
    v4u a, b;
};
__device__ __forceinline__ v16bf frag_cast(const Frag& f) {
    return __builtin_bit_cast(v16bf, f);
}

// ---------------------------------------------------------------------------
// CDNA5 async global->LDS copy (ASYNCcnt-tracked, bypasses VGPRs).
// Probe-learned signature: (int4 AS1* global_src, int4 AS3* lds_dst, Ii, Ii).
// ---------------------------------------------------------------------------
#if defined(__has_builtin)
#  if __has_builtin(__builtin_amdgcn_global_load_async_to_lds_b128)
#    define HAVE_ASYNC_LDS 1
#  endif
#  if __has_builtin(__builtin_amdgcn_s_wait_asynccnt)
#    define HAVE_WAIT_ASYNC 1
#  endif
#endif

#define AS_GLOBAL __attribute__((address_space(1)))
#define AS_LDS    __attribute__((address_space(3)))

__device__ __forceinline__ void async_copy16(const unsigned short* g,
                                             unsigned short* l) {
#if defined(HAVE_ASYNC_LDS)
    __builtin_amdgcn_global_load_async_to_lds_b128(
        (AS_GLOBAL v4i_async*)g, (AS_LDS v4i_async*)l, 0, 0);
#else
    *(v4u*)l = *(const v4u*)g;
#endif
}

__device__ __forceinline__ void wait_async0() {
#if defined(HAVE_WAIT_ASYNC)
    __builtin_amdgcn_s_wait_asynccnt(0);
#elif defined(HAVE_ASYNC_LDS)
    asm volatile("s_wait_asynccnt 0" ::: "memory");
#endif
}

// ---------------------------------------------------------------------------
// Native bf16 conversions (RNE) -> v_cvt_pk_bf16_f32-class hardware ops.
// ---------------------------------------------------------------------------
__device__ __forceinline__ unsigned short f32_to_bf16(float f) {
    return __builtin_bit_cast(unsigned short, (__bf16)f);
}
__device__ __forceinline__ float bf16_to_f32(unsigned short h) {
    return __uint_as_float(((unsigned)h) << 16);
}
struct HiLo {
    unsigned short hi, lo;
};
__device__ __forceinline__ HiLo split_hi_lo(float x) {
    __bf16 h = (__bf16)x;                 // RNE truncate
    __bf16 l = (__bf16)(x - (float)h);    // residual, ~2^-9 relative
    HiLo r;
    r.hi = __builtin_bit_cast(unsigned short, h);
    r.lo = __builtin_bit_cast(unsigned short, l);
    return r;
}

// ---------------------------------------------------------------------------
// Reference fake-FP8 quantizer, computed exactly (no transcendentals):
//   e  = clip(floor(log2(|a|+1e-8)), -7, 7)   -> exponent field extraction
//   mq = rne(( |a|*2^-e - 1 ) * 8) / 8        -> v_rndne (round half to even)
//   q  = sign * (1 + mq) * 2^e
// Every output has <=4 significant mantissa bits -> exactly representable in bf16.
// ---------------------------------------------------------------------------
__device__ __forceinline__ float q8(float a) {
    float ax = __builtin_fabsf(a);
    float y  = ax + 1e-8f;                       // > 0, normal
    int   E  = (int)((__float_as_uint(y) >> 23) & 0xFFu) - 127;  // exact floor(log2 y)
    E = E > 7 ? 7 : (E < -7 ? -7 : E);
    float p    = __uint_as_float((unsigned)(E + 127) << 23);     // 2^e
    float pinv = __uint_as_float((unsigned)(127 - E) << 23);     // 2^-e (exact)
    float m    = ax * pinv - 1.0f;
    float mq   = __builtin_rintf(m * 8.0f) * 0.125f;
    float v    = (1.0f + mq) * p;
    return a < 0.0f ? -v : v;                    // a==0 -> v==0 either way
}

// ---------------------------------------------------------------------------
// Phase 1: elementwise quantize f32 weights -> bf16 (exact)
// ---------------------------------------------------------------------------
__global__ __launch_bounds__(256) void quantize_bf16_kernel(
    const float* __restrict__ in, unsigned short* __restrict__ out, int n) {
    int i = (blockIdx.x * 256 + threadIdx.x) * 4;
    if (i + 3 < n) {
        v4f v = *(const v4f*)(in + i);
        v4us o;
        o.x = f32_to_bf16(q8(v.x));
        o.y = f32_to_bf16(q8(v.y));
        o.z = f32_to_bf16(q8(v.z));
        o.w = f32_to_bf16(q8(v.w));
        *(v4us*)(out + i) = o;
    }
}

// ---------------------------------------------------------------------------
// Tiled WMMA GEMM with q8 epilogue, double-buffered LDS pipeline.
//   out[i][j] = q8( sum_k L[i][k] * W[j][k] )        (W plays the role of B^T)
// Block tile 128(M) x 128(N), K-step 32, 256 threads = 8 waves in a 2x4 grid,
// each wave owns a 64x32 sub-tile = 4x2 accumulators of 16x16.
// Weight tiles (and bf16 left tiles) are staged with async global->LDS copies;
// the f32 left tile (SPLIT) flows global->VGPR->(hi/lo bf16)->LDS, with the
// global loads issued before the WMMA block and the stores after it.
// ---------------------------------------------------------------------------
#define LDS_STRIDE 40   // ushorts per row: 80B -> all b128 frag loads 16B aligned

template <bool SPLIT>
__global__ __launch_bounds__(256) void gemm_q8_kernel(
    const void* __restrict__ Lseg0, const void* __restrict__ Lseg1,
    const unsigned short* __restrict__ Wseg0,
    const unsigned short* __restrict__ Wseg1,
    int ldL0, int ldL1, int ldW0, int ldW1,
    int Kseg0, int Ktot,
    float* __restrict__ outF, unsigned short* __restrict__ outB, int ldOut) {

    __shared__ __align__(16) unsigned short sAhi[2][128 * LDS_STRIDE];
    __shared__ __align__(16) unsigned short sAlo[SPLIT ? 2 : 1][128 * LDS_STRIDE];
    __shared__ __align__(16) unsigned short sBt [2][128 * LDS_STRIDE];

    const int tid   = threadIdx.x;
    const int wave  = tid >> 5;          // 0..7
    const int lane  = tid & 31;          // wave32
    const int cpart = lane >> 4;         // lane half: selects K sub-ranges
    const int mrow  = lane & 15;

    const int waveM = wave >> 2;         // 0..1
    const int waveN = wave & 3;          // 0..3
    const int rw    = waveM * 64;        // wave row offset in block tile
    const int cw    = waveN * 32;        // wave col offset in block tile

    const int N0 = blockIdx.x * 128;
    const int M0 = blockIdx.y * 128;

    v8f acc[4][2];
#pragma unroll
    for (int i = 0; i < 4; ++i)
#pragma unroll
        for (int j = 0; j < 2; ++j)
            acc[i][j] = (v8f){0.f, 0.f, 0.f, 0.f, 0.f, 0.f, 0.f, 0.f};

    v4f Lreg[4];     // in-flight f32 left tile (SPLIT path)

    // ---- staging helpers -------------------------------------------------
    auto stage_w = [&](int kt, int b) {     // 128 x 32 bf16 weight tile, async
        const bool seg1 = (kt >= Kseg0);
        const int  kk   = seg1 ? kt - Kseg0 : kt;
        const unsigned short* W  = seg1 ? Wseg1 : Wseg0;
        const int             ldW = seg1 ? ldW1 : ldW0;
        int t = tid;
#pragma unroll
        for (int it = 0; it < 2; ++it, t += 256) {
            int row = t >> 2, q = t & 3;
            async_copy16(W + (size_t)(N0 + row) * ldW + kk + q * 8,
                         &sBt[b][row * LDS_STRIDE + q * 8]);
        }
    };
    auto stage_left_bf16 = [&](int kt, int b) {   // !SPLIT: bf16 left, async
        const unsigned short* L = (const unsigned short*)Lseg0;
        int t = tid;
#pragma unroll
        for (int it = 0; it < 2; ++it, t += 256) {
            int row = t >> 2, q = t & 3;
            async_copy16(L + (size_t)(M0 + row) * ldL0 + kt + q * 8,
                         &sAhi[b][row * LDS_STRIDE + q * 8]);
        }
    };
    auto load_left_f32 = [&](int kt) {            // SPLIT: issue global loads
        const bool  seg1 = (kt >= Kseg0);
        const int   kk   = seg1 ? kt - Kseg0 : kt;
        const float* L   = seg1 ? (const float*)Lseg1 : (const float*)Lseg0;
        const int    ldL = seg1 ? ldL1 : ldL0;
        int t = tid;
#pragma unroll
        for (int it = 0; it < 4; ++it, t += 256) {
            int row = t >> 3, cch = t & 7;
            Lreg[it] = *(const v4f*)(L + (size_t)(M0 + row) * ldL + kk + cch * 4);
        }
    };
    auto store_left_split = [&](int b) {          // SPLIT: convert + LDS store
        int t = tid;
#pragma unroll
        for (int it = 0; it < 4; ++it, t += 256) {
            int row = t >> 3, cch = t & 7;
            v4f v = Lreg[it];
            v4us hi, lo;
            HiLo sx = split_hi_lo(v.x); hi.x = sx.hi; lo.x = sx.lo;
            HiLo sy = split_hi_lo(v.y); hi.y = sy.hi; lo.y = sy.lo;
            HiLo sz = split_hi_lo(v.z); hi.z = sz.hi; lo.z = sz.lo;
            HiLo sw = split_hi_lo(v.w); hi.w = sw.hi; lo.w = sw.lo;
            *(v4us*)&sAhi[b][row * LDS_STRIDE + cch * 4] = hi;
            *(v4us*)&sAlo[b][row * LDS_STRIDE + cch * 4] = lo;
        }
    };

    // ---- prologue: stage tile 0 into buffer 0 ----------------------------
    stage_w(0, 0);
    if (SPLIT) {
        load_left_f32(0);
        store_left_split(0);
    } else {
        stage_left_bf16(0, 0);
    }
    wait_async0();
    __syncthreads();

    // ---- pipelined main loop: one barrier per K-step ---------------------
    int buf = 0;
    for (int kt = 0; kt < Ktot; kt += 32) {
        const int  ktn  = kt + 32;
        const bool more = ktn < Ktot;

        if (more) {                       // issue next tile's traffic first
            stage_w(ktn, buf ^ 1);
            if (SPLIT) load_left_f32(ktn);
            else       stage_left_bf16(ktn, buf ^ 1);
        }

        // -- preload ALL fragments so the DS pipe runs ahead of the WMMAs
        Frag bfr[2];
#pragma unroll
        for (int nn = 0; nn < 2; ++nn) {
            const unsigned short* p =
                &sBt[buf][(cw + nn * 16 + mrow) * LDS_STRIDE + cpart * 8];
            bfr[nn].a = *(const v4u*)p;
            bfr[nn].b = *(const v4u*)(p + 16);
        }
        Frag ah[4], al[4];
#pragma unroll
        for (int mm = 0; mm < 4; ++mm) {
            const unsigned short* pa =
                &sAhi[buf][(rw + mm * 16 + mrow) * LDS_STRIDE];
            ah[mm].a = *(const v4u*)(pa + 8 * cpart);       // K = 8c..8c+7
            ah[mm].b = *(const v4u*)(pa + 16 + 8 * cpart);  // K = 16+8c..+7
            if (SPLIT) {
                const unsigned short* pl =
                    &sAlo[buf][(rw + mm * 16 + mrow) * LDS_STRIDE];
                al[mm].a = *(const v4u*)(pl + 8 * cpart);
                al[mm].b = *(const v4u*)(pl + 16 + 8 * cpart);
            }
        }
        // -- WMMA chain
#pragma unroll
        for (int mm = 0; mm < 4; ++mm) {
#pragma unroll
            for (int nn = 0; nn < 2; ++nn) {
                acc[mm][nn] = __builtin_amdgcn_wmma_f32_16x16x32_bf16(
                    false, frag_cast(ah[mm]), false, frag_cast(bfr[nn]),
                    (short)0, acc[mm][nn], false, false);
                if (SPLIT) {
                    acc[mm][nn] = __builtin_amdgcn_wmma_f32_16x16x32_bf16(
                        false, frag_cast(al[mm]), false, frag_cast(bfr[nn]),
                        (short)0, acc[mm][nn], false, false);
                }
            }
        }

        if (more && SPLIT) store_left_split(buf ^ 1);  // convert after compute
        wait_async0();
        __syncthreads();
        buf ^= 1;
    }

    // ---- q8 epilogue; C layout: VGPR v -> row v + 8*cpart, col = lane&15
#pragma unroll
    for (int mm = 0; mm < 4; ++mm) {
#pragma unroll
        for (int nn = 0; nn < 2; ++nn) {
#pragma unroll
            for (int v = 0; v < 8; ++v) {
                int rg = M0 + rw + mm * 16 + 8 * cpart + v;
                int cg = N0 + cw + nn * 16 + mrow;
                float val = q8(acc[mm][nn][v]);
                outF[(size_t)rg * ldOut + cg] = val;
                if (outB) outB[(size_t)rg * ldOut + cg] = f32_to_bf16(val);
            }
        }
    }
}

// ---------------------------------------------------------------------------
// Host launcher
// ---------------------------------------------------------------------------
extern "C" void kernel_launch(void* const* d_in, const int* in_sizes, int n_in,
                              void* d_out, int out_size, void* d_ws, size_t ws_size,
                              hipStream_t stream) {
    constexpr int BATCH = 4096, N = 2048, M = 1024, P = 1024;

    const float* x = (const float*)d_in[0];   // 4096 x 2048
    const float* u = (const float*)d_in[1];   // 4096 x 1024
    const float* A = (const float*)d_in[2];   // 2048 x 2048
    const float* B = (const float*)d_in[3];   // 2048 x 1024
    const float* C = (const float*)d_in[4];   // 1024 x 2048

    // Workspace layout (bf16): Aq | Bq | Cq | x_next   (total 32 MB)
    unsigned short* Aq = (unsigned short*)d_ws;
    unsigned short* Bq = Aq + (size_t)N * N;
    unsigned short* Cq = Bq + (size_t)N * M;
    unsigned short* Xq = Cq + (size_t)P * N;

    float* xnext = (float*)d_out;                       // 4096 x 2048 f32
    float* y     = xnext + (size_t)BATCH * N;           // 4096 x 1024 f32

    // Phase 1: quantize weights to bf16 (exact representation of q8 grid)
    quantize_bf16_kernel<<<(N * N) / 1024, 256, 0, stream>>>(A, Aq, N * N);
    quantize_bf16_kernel<<<(N * M) / 1024, 256, 0, stream>>>(B, Bq, N * M);
    quantize_bf16_kernel<<<(P * N) / 1024, 256, 0, stream>>>(C, Cq, P * N);

    // Phase 2: x_next = q8( x@Aq^T + u@Bq^T ), fused over K = 2048+1024,
    //          hi/lo bf16 split on x,u for ~f32 accuracy.
    gemm_q8_kernel<true><<<dim3(N / 128, BATCH / 128), 256, 0, stream>>>(
        x, u, Aq, Bq,
        /*ldL0*/ N, /*ldL1*/ M, /*ldW0*/ N, /*ldW1*/ M,
        /*Kseg0*/ N, /*Ktot*/ N + M,
        xnext, Xq, /*ldOut*/ N);

    // Phase 3: y = q8( x_next @ Cq^T ), x_next already exact in bf16.
    gemm_q8_kernel<false><<<dim3(P / 128, BATCH / 128), 256, 0, stream>>>(
        Xq, nullptr, Cq, nullptr,
        /*ldL0*/ N, /*ldL1*/ 0, /*ldW0*/ N, /*ldW1*/ 0,
        /*Kseg0*/ N, /*Ktot*/ N,
        y, nullptr, /*ldOut*/ P);
}
